// Net_59038620450860
// MI455X (gfx1250) — matrix operand order repaired and checked by reading, not compiled, not run on hardware
//
#include <hip/hip_runtime.h>
#include <hip/hip_bf16.h>

typedef __attribute__((ext_vector_type(16))) _Float16 v16h;
typedef __attribute__((ext_vector_type(8)))  _Float16 v8h;
typedef __attribute__((ext_vector_type(4)))  _Float16 v4h;
typedef __attribute__((ext_vector_type(8)))  float    v8f;

#define B_   256
#define T_   2048
#define F1   128      // input features, layer 1
#define H_   64       // hidden
#define NG   256      // 4*H gate columns
#define SA1  200      // LDS stride (halves) for layer-1 A buffer (padded, 16B-aligned rows)
#define SA2  136      // LDS stride (halves) for layer-2 A buffer
#define BM   16       // batch rows per workgroup

#define LOG2E 1.4426950408889634f

// --- fast transcendentals (single v_exp_f32 / v_rcp_f32 / v_tanh_f32) ------
__device__ __forceinline__ float fast_sigm(float x) {
    float e = __builtin_amdgcn_exp2f(-x * LOG2E);
    return __builtin_amdgcn_rcpf(1.0f + e);
}

__device__ __forceinline__ float fast_tanh(float x) {
#if __has_builtin(__builtin_amdgcn_tanhf)
    return __builtin_amdgcn_tanhf(x);
#else
    float e = __builtin_amdgcn_exp2f(x * (-2.0f * LOG2E));
    return 2.0f * __builtin_amdgcn_rcpf(1.0f + e) - 1.0f;
#endif
}

// --- WMMA fragment loaders -------------------------------------------------
// A (f16 16x32, MxK) per ISA layout: lanes 0-15 row=lane, K halves {0..7,16..23};
// lanes 16-31 row=lane-16, K halves {8..15,24..31}. Two contiguous 16B runs.
__device__ __forceinline__ v16h load_a_frag(const _Float16* sA, int stride, int kc, int lane) {
    int row = lane & 15;
    int klo = (lane >= 16) ? 8 : 0;
    const _Float16* p = sA + row * stride + kc * 32 + klo;
    v8h lo = *(const v8h*)(p);
    v8h hi = *(const v8h*)(p + 16);
    return __builtin_shufflevector(lo, hi, 0,1,2,3,4,5,6,7,8,9,10,11,12,13,14,15);
}

// B (f16 32x16, KxN): lane 0-15 -> col=lane, K = kbase+{0..7,16..23};
// lane 16-31 -> col=lane-16, kbase shifted by +8.  Combined weight = [W_ih ; W_hh].
__device__ __forceinline__ v16h load_b_frag(const float* __restrict__ Wih,
                                            const float* __restrict__ Whh,
                                            int Fin, int n, int kbase) {
    v16h b;
#pragma unroll
    for (int e = 0; e < 16; ++e) {
        int k = kbase + (e < 8 ? e : e + 8);
        float v = (k < Fin) ? Wih[n * Fin + k] : Whh[n * H_ + (k - Fin)];
        b[e] = (_Float16)v;
    }
    return b;
}

// --- Persistent fused 2-layer LSTM + head ----------------------------------
__global__ __launch_bounds__(256) void lstm2_fused_kernel(
    const float* __restrict__ x,
    const float* __restrict__ W_ih1, const float* __restrict__ W_hh1,
    const float* __restrict__ b_ih1, const float* __restrict__ b_hh1,
    const float* __restrict__ W_ih2, const float* __restrict__ W_hh2,
    const float* __restrict__ b_ih2, const float* __restrict__ b_hh2,
    const float* __restrict__ W_out, const float* __restrict__ b_out,
    float* __restrict__ out)
{
    __shared__ __align__(16) _Float16 sA1[BM * SA1];   // [x_t (128) | h1 (64)] per row
    __shared__ __align__(16) _Float16 sA2[BM * SA2];   // [h1 (64)   | h2 (64)] per row
    __shared__ __align__(16) float    sG[BM * NG];     // activated gates
    __shared__ float                  sWout[H_];

    const int tid  = threadIdx.x;
    const int lane = tid & 31;
    const int w    = tid >> 5;           // wave id, 0..7 -> gate cols w*32..w*32+31
    const int b0   = blockIdx.x * BM;
    const int hi8  = (lane >= 16) ? 8 : 0;
    // gate type 0=i,1=f,2=g,3=o -- wave-uniform: force into SGPR for scalar branch
    const int gtyp = __builtin_amdgcn_readfirstlane(w >> 1);

    // ---- load all weights into resident WMMA B-fragments (registers) ----
    v16h B1[6][2], B2[4][2];
    float bias1[2], bias2[2];
    {
        const int kadd = (lane >= 16) ? 8 : 0;
#pragma unroll
        for (int nt = 0; nt < 2; ++nt) {
            const int n = (w << 5) + (nt << 4) + (lane & 15);
            bias1[nt] = b_ih1[n] + b_hh1[n];
            bias2[nt] = b_ih2[n] + b_hh2[n];
#pragma unroll
            for (int kc = 0; kc < 6; ++kc)
                B1[kc][nt] = load_b_frag(W_ih1, W_hh1, F1, n, kc * 32 + kadd);
#pragma unroll
            for (int kc = 0; kc < 4; ++kc)
                B2[kc][nt] = load_b_frag(W_ih2, W_hh2, H_, n, kc * 32 + kadd);
        }
    }
    if (tid < H_) sWout[tid] = W_out[tid];
    const float bout = b_out[0];

    // ---- init state: h=0 in LDS, c=0 in registers ----
    for (int i = tid; i < BM * SA1; i += 256) sA1[i] = (_Float16)0.0f;
    for (int i = tid; i < BM * SA2; i += 256) sA2[i] = (_Float16)0.0f;
    float c1[4] = {0.f, 0.f, 0.f, 0.f};
    float c2[4] = {0.f, 0.f, 0.f, 0.f};

    const int urow = tid >> 4;           // elementwise-update row (0..15)
    const int ucol = (tid & 15) * 4;     // 4 consecutive hidden cols
    const int xrow = tid >> 4;           // x-load row
    const int xf   = (tid & 15) * 8;     // 8 consecutive features
    __syncthreads();

    // ---- prime x pipeline: registers hold x tile for step t ----
    const float* xp0 = x + ((size_t)(b0 + xrow) * T_) * F1 + xf;
    float4 xa = *(const float4*)xp0;
    float4 xb = *(const float4*)(xp0 + 4);

#pragma unroll 1
    for (int t = 0; t < T_; ++t) {
        // ---- stage current x tile (already in registers) into LDS as f16 ----
        {
            v8h hv;
            hv[0] = (_Float16)xa.x; hv[1] = (_Float16)xa.y;
            hv[2] = (_Float16)xa.z; hv[3] = (_Float16)xa.w;
            hv[4] = (_Float16)xb.x; hv[5] = (_Float16)xb.y;
            hv[6] = (_Float16)xb.z; hv[7] = (_Float16)xb.w;
            *(v8h*)&sA1[xrow * SA1 + xf] = hv;
        }
        // ---- issue prefetch of x for step t+1 (latency hides under this step) ----
        float4 na, nb;
        {
            const int tn = (t + 1 < T_) ? (t + 1) : t;   // clamp, avoids OOB
            const float* xpn = x + ((size_t)(b0 + xrow) * T_ + (size_t)tn) * F1 + xf;
            na = *(const float4*)xpn;
            nb = *(const float4*)(xpn + 4);
        }
        __syncthreads();

        // ---- layer 1 gates: [16,192] x [192,256] via WMMA (1-deep A pipeline) ----
        {
            v8f acc[2];
#pragma unroll
            for (int r = 0; r < 8; ++r) { acc[0][r] = bias1[0]; acc[1][r] = bias1[1]; }
            v16h acur = load_a_frag(sA1, SA1, 0, lane);
#pragma unroll
            for (int kc = 0; kc < 5; ++kc) {
                v16h anext = load_a_frag(sA1, SA1, kc + 1, lane);
                acc[0] = __builtin_amdgcn_wmma_f32_16x16x32_f16(false, acur, false, B1[kc][0],
                                                                (short)0, acc[0], false, false);
                acc[1] = __builtin_amdgcn_wmma_f32_16x16x32_f16(false, acur, false, B1[kc][1],
                                                                (short)0, acc[1], false, false);
                acur = anext;
            }
            acc[0] = __builtin_amdgcn_wmma_f32_16x16x32_f16(false, acur, false, B1[5][0],
                                                            (short)0, acc[0], false, false);
            acc[1] = __builtin_amdgcn_wmma_f32_16x16x32_f16(false, acur, false, B1[5][1],
                                                            (short)0, acc[1], false, false);
#pragma unroll
            for (int nt = 0; nt < 2; ++nt) {
                const int n = (w << 5) + (nt << 4) + (lane & 15);
                if (gtyp == 2) {
#pragma unroll
                    for (int r = 0; r < 8; ++r)
                        sG[(r + hi8) * NG + n] = fast_tanh(acc[nt][r]);
                } else {
#pragma unroll
                    for (int r = 0; r < 8; ++r)
                        sG[(r + hi8) * NG + n] = fast_sigm(acc[nt][r]);
                }
            }
        }
        __syncthreads();

        // ---- c1/h1 elementwise update; h1 -> sA1 (next step) and sA2 (layer-2 input) ----
        {
            const int gb = urow * NG + ucol;
            float4 gi = *(const float4*)&sG[gb];
            float4 gf = *(const float4*)&sG[gb + 64];
            float4 gg = *(const float4*)&sG[gb + 128];
            float4 go = *(const float4*)&sG[gb + 192];
            float iv[4] = {gi.x, gi.y, gi.z, gi.w};
            float fv[4] = {gf.x, gf.y, gf.z, gf.w};
            float gv[4] = {gg.x, gg.y, gg.z, gg.w};
            float ov[4] = {go.x, go.y, go.z, go.w};
            v4h hv;
#pragma unroll
            for (int j = 0; j < 4; ++j) {
                c1[j] = fv[j] * c1[j] + iv[j] * gv[j];
                float h = ov[j] * fast_tanh(c1[j]);
                hv[j] = (_Float16)h;
            }
            *(v4h*)&sA1[urow * SA1 + F1 + ucol] = hv;
            *(v4h*)&sA2[urow * SA2 + ucol]      = hv;
        }
        __syncthreads();

        // ---- layer 2 gates: [16,128] x [128,256] (1-deep A pipeline) ----
        {
            v8f acc[2];
#pragma unroll
            for (int r = 0; r < 8; ++r) { acc[0][r] = bias2[0]; acc[1][r] = bias2[1]; }
            v16h acur = load_a_frag(sA2, SA2, 0, lane);
#pragma unroll
            for (int kc = 0; kc < 3; ++kc) {
                v16h anext = load_a_frag(sA2, SA2, kc + 1, lane);
                acc[0] = __builtin_amdgcn_wmma_f32_16x16x32_f16(false, acur, false, B2[kc][0],
                                                                (short)0, acc[0], false, false);
                acc[1] = __builtin_amdgcn_wmma_f32_16x16x32_f16(false, acur, false, B2[kc][1],
                                                                (short)0, acc[1], false, false);
                acur = anext;
            }
            acc[0] = __builtin_amdgcn_wmma_f32_16x16x32_f16(false, acur, false, B2[3][0],
                                                            (short)0, acc[0], false, false);
            acc[1] = __builtin_amdgcn_wmma_f32_16x16x32_f16(false, acur, false, B2[3][1],
                                                            (short)0, acc[1], false, false);
#pragma unroll
            for (int nt = 0; nt < 2; ++nt) {
                const int n = (w << 5) + (nt << 4) + (lane & 15);
                if (gtyp == 2) {
#pragma unroll
                    for (int r = 0; r < 8; ++r)
                        sG[(r + hi8) * NG + n] = fast_tanh(acc[nt][r]);
                } else {
#pragma unroll
                    for (int r = 0; r < 8; ++r)
                        sG[(r + hi8) * NG + n] = fast_sigm(acc[nt][r]);
                }
            }
        }
        __syncthreads();

        // ---- c2/h2 update + fused output head (shuffle reduction, no barrier) ----
        {
            const int gb = urow * NG + ucol;
            float4 gi = *(const float4*)&sG[gb];
            float4 gf = *(const float4*)&sG[gb + 64];
            float4 gg = *(const float4*)&sG[gb + 128];
            float4 go = *(const float4*)&sG[gb + 192];
            float iv[4] = {gi.x, gi.y, gi.z, gi.w};
            float fv[4] = {gf.x, gf.y, gf.z, gf.w};
            float gv[4] = {gg.x, gg.y, gg.z, gg.w};
            float ov[4] = {go.x, go.y, go.z, go.w};
            v4h hv;
            float partial = 0.0f;
#pragma unroll
            for (int j = 0; j < 4; ++j) {
                c2[j] = fv[j] * c2[j] + iv[j] * gv[j];
                float h = ov[j] * fast_tanh(c2[j]);
                hv[j] = (_Float16)h;
                partial += h * sWout[ucol + j];
            }
            *(v4h*)&sA2[urow * SA2 + H_ + ucol] = hv;

            // rows occupy 16 contiguous lanes of a wave32: butterfly-reduce within 16
            partial += __shfl_xor(partial, 1);
            partial += __shfl_xor(partial, 2);
            partial += __shfl_xor(partial, 4);
            partial += __shfl_xor(partial, 8);
            if ((tid & 15) == 0) {
                float s = fminf(fmaxf(partial + bout, 0.0f), 1.0f);
                out[(size_t)(b0 + urow) * T_ + (size_t)t] = s;
            }
        }
        // rotate x pipeline registers
        xa = na;
        xb = nb;
        // no loop-end barrier needed: next x-store touches a region whose readers
        // all retired before the post-gates1 barrier of this iteration.
    }
}

extern "C" void kernel_launch(void* const* d_in, const int* in_sizes, int n_in,
                              void* d_out, int out_size, void* d_ws, size_t ws_size,
                              hipStream_t stream) {
    (void)in_sizes; (void)n_in; (void)out_size; (void)d_ws; (void)ws_size;
    const float* x     = (const float*)d_in[0];
    const float* W_ih1 = (const float*)d_in[1];
    const float* W_hh1 = (const float*)d_in[2];
    const float* b_ih1 = (const float*)d_in[3];
    const float* b_hh1 = (const float*)d_in[4];
    const float* W_ih2 = (const float*)d_in[5];
    const float* W_hh2 = (const float*)d_in[6];
    const float* b_ih2 = (const float*)d_in[7];
    const float* b_hh2 = (const float*)d_in[8];
    const float* W_out = (const float*)d_in[9];
    const float* b_out = (const float*)d_in[10];

    lstm2_fused_kernel<<<dim3(B_ / BM), dim3(256), 0, stream>>>(
        x, W_ih1, W_hh1, b_ih1, b_hh1,
        W_ih2, W_hh2, b_ih2, b_hh2,
        W_out, b_out, (float*)d_out);
}
